// SCA_ConvBlock_5136780886740
// MI455X (gfx1250) — compile-verified, matrix operand order checked by reading, not run on hardware
//
#include <hip/hip_runtime.h>
#include <cstddef>

typedef _Float16 v8h  __attribute__((ext_vector_type(8)));
typedef _Float16 v16h __attribute__((ext_vector_type(16)));
typedef float    v8f  __attribute__((ext_vector_type(8)));

#define B_   8
#define C_   64
#define H_   256
#define W_   256
#define HW_  (H_*W_)
#define CHW_ (C_*HW_)

// ---- WMMA conv tile geometry ----
#define WTILE  128        // pixels per workgroup tile (one row segment)
#define XROW   130        // WTILE + 2 halo columns
#define XPITCH 72         // halfs per (row,wl) across ci: 64 + 8 pad (16B multiple, 36-dword bank step)
#define KDIM   576        // Cin * 3 * 3
#define KSTEPS 18         // 576 / 32

// =====================================================================
// Pass 0: pre-convert wv to f16 in workspace, K reordered as (khw, ci) --
// exactly the A-fragment layout.  Shared by all conv blocks (L2-resident).
// =====================================================================
__global__ __launch_bounds__(256)
void k_wprep(const float* __restrict__ wv, _Float16* __restrict__ wh) {
  const int idx = blockIdx.x * 256 + threadIdx.x;     // 0..36863
  if (idx >= C_ * KDIM) return;
  const int co  = idx / KDIM;
  const int r   = idx - co * KDIM;                    // khw*64 + ci
  const int khw = r >> 6;
  const int ci  = r & 63;
  const int kh  = khw / 3;
  const int kw  = khw - kh * 3;
  wh[idx] = (_Float16)wv[((co * C_ + ci) * 3 + kh) * 3 + kw];
}

// =====================================================================
// Pass A: per-(b,c,w) column statistics  Sx = sum_h x, x0 = x[h=0], xL = x[h=H-1]
// =====================================================================
__global__ __launch_bounds__(256)
void k_colstats(const float* __restrict__ x,
                float* __restrict__ Sx, float* __restrict__ x0, float* __restrict__ xL) {
  const int bc = blockIdx.x;              // b*C + c, 0..511
  const int w  = threadIdx.x;             // 0..255
  const float* col = x + (size_t)bc * HW_ + w;
  float s = 0.0f;
  #pragma unroll 8
  for (int h = 0; h < H_; ++h) s += col[(size_t)h * W_];
  Sx[bc * W_ + w] = s;
  x0[bc * W_ + w] = col[0];
  xL[bc * W_ + w] = col[(size_t)(H_ - 1) * W_];
}

// =====================================================================
// Pass B1: partial dot(sq, sk) per (b, co, w-chunk of 16).
// sq[w] = sum_{ci,kh,kw} wq[co,ci,kh,kw] * A_kh[ci, w+kw-1]
// with A0 = Sx - xL, A1 = Sx, A2 = Sx - x0  (exact column-sum identity).
// =====================================================================
__global__ __launch_bounds__(256)
void k_partial(const float* __restrict__ Sx, const float* __restrict__ x0,
               const float* __restrict__ xL, const float* __restrict__ wq,
               const float* __restrict__ wk, float* __restrict__ fpart) {
  __shared__ float As[3][C_][20];          // 18 used, padded
  __shared__ float red[256];
  const int wc = blockIdx.x;               // 0..15
  const int b  = blockIdx.y;               // 0..7
  const int w0 = wc * 16;
  const int t  = threadIdx.x;

  for (int idx = t; idx < 3 * C_ * 18; idx += 256) {
    int kh = idx / (C_ * 18);
    int r  = idx - kh * (C_ * 18);
    int ci = r / 18;
    int j  = r - ci * 18;
    int w  = w0 - 1 + j;
    float a = 0.0f;
    if ((unsigned)w < (unsigned)W_) {
      int o = (b * C_ + ci) * W_ + w;
      float s = Sx[o];
      a = (kh == 0) ? (s - xL[o]) : ((kh == 1) ? s : (s - x0[o]));
    }
    As[kh][ci][j] = a;
  }
  __syncthreads();

  const int co = t & 63;
  const int wg = t >> 6;                   // 4 w's per thread
  float aq[4] = {0.f, 0.f, 0.f, 0.f};
  float ak[4] = {0.f, 0.f, 0.f, 0.f};
  const float* wqp = wq + (size_t)co * KDIM;   // [co][ci][kh][kw]
  const float* wkp = wk + (size_t)co * KDIM;
  for (int ci = 0; ci < C_; ++ci) {
    #pragma unroll
    for (int khw = 0; khw < 9; ++khw) {
      const int kh = khw / 3;
      const int kw = khw - kh * 3;
      const float cq = wqp[ci * 9 + khw];
      const float ck = wkp[ci * 9 + khw];
      #pragma unroll
      for (int j = 0; j < 4; ++j) {
        const float a = As[kh][ci][wg * 4 + j + kw];   // broadcast LDS read
        aq[j] = fmaf(cq, a, aq[j]);
        ak[j] = fmaf(ck, a, ak[j]);
      }
    }
  }
  float partial = 0.0f;
  #pragma unroll
  for (int j = 0; j < 4; ++j) partial += aq[j] * ak[j];
  red[t] = partial;
  __syncthreads();
  if (t < 64) {
    float sum = red[t] + red[t + 64] + red[t + 128] + red[t + 192];
    fpart[(b * 16 + wc) * C_ + t] = sum * (1.0f / 16.0f);   // / sqrt(W)
  }
}

// =====================================================================
// Pass B2: softmax over channels -> scores; block 0 also precomputes BN inv/bias.
// =====================================================================
__global__ __launch_bounds__(64)
void k_softmax(const float* __restrict__ fpart, const float* __restrict__ gamma,
               const float* __restrict__ beta, const float* __restrict__ rmean,
               const float* __restrict__ rvar, float* __restrict__ scores,
               float* __restrict__ invA, float* __restrict__ biasA) {
  __shared__ float fs[C_];
  const int b = blockIdx.x;
  const int t = threadIdx.x;               // 0..63
  float f = 0.0f;
  for (int wc = 0; wc < 16; ++wc) f += fpart[(b * 16 + wc) * C_ + t];
  fs[t] = f;
  __syncthreads();
  float m = -3.0e38f;
  for (int i = 0; i < C_; ++i) m = fmaxf(m, fs[i]);
  float den = 0.0f;
  for (int i = 0; i < C_; ++i) den += __expf(fs[i] - m);
  scores[b * C_ + t] = __expf(f - m) / den;
  if (b == 0) {
    float iv = gamma[t] * rsqrtf(rvar[t] + 1e-5f);
    invA[t]  = iv;
    biasA[t] = beta[t] - rmean[t] * iv;
  }
}

// =====================================================================
// Pass C: fv = conv3x3(x, wv) via WMMA implicit GEMM (M=64 co, K=576, N=128 px),
// fused epilogue: relu((scores*fv + x) * inv + bias).
// A fragments stream from the f16 weight image in global (LOADcnt, L0/L2);
// B fragments from the transposed f16 activation tile in LDS (DScnt).
// Explicit 1-deep software pipeline: fragments for k-step j+1 are issued
// before the 4 WMMAs of step j, so waits are graduated instead of wait-0.
// =====================================================================
__global__ __launch_bounds__(256)
void k_conv_fused(const float* __restrict__ x, const _Float16* __restrict__ wh,
                  const float* __restrict__ scores, const float* __restrict__ invA,
                  const float* __restrict__ biasA, float* __restrict__ out) {
  extern __shared__ _Float16 smem[];
  _Float16* Xl = smem;                      // [(row*XROW+wl)][XPITCH] ci-contiguous

  const int tid = threadIdx.x;
  const int wc  = blockIdx.x;               // 0..1
  const int h0  = blockIdx.y;               // 0..255
  const int b   = blockIdx.z;               // 0..7
  const int w0  = wc * WTILE;

  // ---- stage activations rows h0-1..h0+1, w0-1..w0+WTILE (transposed: ci innermost) ----
  // lane-consecutive wl => coalesced global reads; pitch-72 LDS stores walk a
  // 36-dword bank step (conflict-free across a half-wave).
  for (int idx = tid; idx < 3 * C_ * XROW; idx += 256) {
    int row = idx / (C_ * XROW);
    int r   = idx - row * (C_ * XROW);
    int ci  = r / XROW;
    int wl  = r - ci * XROW;
    int h   = h0 - 1 + row;
    int w   = w0 - 1 + wl;
    float v = 0.0f;
    if ((unsigned)h < (unsigned)H_ && (unsigned)w < (unsigned)W_)
      v = x[(size_t)b * CHW_ + (size_t)ci * HW_ + (size_t)h * W_ + w];
    Xl[(row * XROW + wl) * XPITCH + ci] = (_Float16)v;
  }
  __syncthreads();

  const int lane  = tid & 31;
  const int wid   = tid >> 5;               // 0..7 waves
  const int mbase = (wid >> 1) * 16;        // co tile base
  const int nbase = (wid & 1) * 64;         // pixel base (4 n-tiles of 16)
  const int ln    = lane & 15;
  const int hi8   = (lane >> 4) << 3;       // K half-select per WMMA A/B layout

  const _Float16* wrow = wh + (size_t)(mbase + ln) * KDIM + hi8;

  // Fragment loaders (j, t become constants after full unroll).
  auto loadA = [&](int j) -> v16h {
    const int khw = j >> 1;
    const int cih = (j & 1) << 5;
    const _Float16* ap = wrow + khw * 64 + cih;       // global (L0/L2)
    v8h lo = *(const v8h*)ap;                         // K = cih + hi8 + 0..7
    v8h hi = *(const v8h*)(ap + 16);                  // K = cih + 16 + hi8 + 0..7
    return __builtin_shufflevector(lo, hi, 0,1,2,3,4,5,6,7,8,9,10,11,12,13,14,15);
  };
  auto loadB = [&](int j, int t) -> v16h {
    const int khw = j >> 1;
    const int cih = (j & 1) << 5;
    const int kh  = khw / 3;
    const int kw  = khw - kh * 3;
    const int p   = nbase + t * 16 + ln;              // output pixel in tile
    const _Float16* bp = Xl + (kh * XROW + p + kw) * XPITCH + cih + hi8;
    v8h lo = *(const v8h*)bp;
    v8h hi = *(const v8h*)(bp + 16);
    return __builtin_shufflevector(lo, hi, 0,1,2,3,4,5,6,7,8,9,10,11,12,13,14,15);
  };

  v8f acc[4] = {};
  v16h a_cur, a_nxt;
  v16h b_cur[4], b_nxt[4];

  // pipeline prologue: fragments for k-step 0
  a_cur = loadA(0);
  #pragma unroll
  for (int t = 0; t < 4; ++t) b_cur[t] = loadB(0, t);

  // K loop: 18 steps of 32 over (kh,kw,ci-half), 1-deep pipelined
  #pragma unroll
  for (int j = 0; j < KSTEPS; ++j) {
    if (j + 1 < KSTEPS) {
      a_nxt = loadA(j + 1);
      #pragma unroll
      for (int t = 0; t < 4; ++t) b_nxt[t] = loadB(j + 1, t);
    }
    #pragma unroll
    for (int t = 0; t < 4; ++t)
      acc[t] = __builtin_amdgcn_wmma_f32_16x16x32_f16(
          false, a_cur, false, b_cur[t], (short)0, acc[t], false, false);
    a_cur = a_nxt;
    #pragma unroll
    for (int t = 0; t < 4; ++t) b_cur[t] = b_nxt[t];
  }

  // ---- fused epilogue: relu((score*fv + x) * inv + bias) ----
  float sc[8], iv[8], bs[8];
  #pragma unroll
  for (int v = 0; v < 8; ++v) {
    const int co = mbase + hi8 + v;         // D layout: VGPR v -> M = hi8 + v
    sc[v] = scores[b * C_ + co];
    iv[v] = invA[co];
    bs[v] = biasA[co];
  }
  #pragma unroll
  for (int t = 0; t < 4; ++t) {
    const int p = nbase + t * 16 + ln;
    const int w = w0 + p;
    #pragma unroll
    for (int v = 0; v < 8; ++v) {
      const int co = mbase + hi8 + v;
      const size_t o = (size_t)b * CHW_ + (size_t)co * HW_ + (size_t)h0 * W_ + w;
      float val = sc[v] * acc[t][v] + x[o];   // exact f32 residual (L2 hit)
      val = val * iv[v] + bs[v];
      // streaming output: non-temporal so the 134MB 'out' stream does not
      // evict the L2-resident x image (192MB L2, x = 134MB, reused 4x).
      __builtin_nontemporal_store(fmaxf(val, 0.0f), &out[o]);
    }
  }
}

// =====================================================================
extern "C" void kernel_launch(void* const* d_in, const int* in_sizes, int n_in,
                              void* d_out, int out_size, void* d_ws, size_t ws_size,
                              hipStream_t stream) {
  (void)in_sizes; (void)n_in; (void)out_size; (void)ws_size;
  const float* x     = (const float*)d_in[0];
  const float* wq    = (const float*)d_in[1];
  const float* wk    = (const float*)d_in[2];
  const float* wv    = (const float*)d_in[3];
  const float* gamma = (const float*)d_in[4];
  const float* beta  = (const float*)d_in[5];
  const float* rmean = (const float*)d_in[6];
  const float* rvar  = (const float*)d_in[7];
  float* out = (float*)d_out;

  float* ws     = (float*)d_ws;
  float* Sx     = ws;                       // 8*64*256 = 131072
  float* x0     = Sx + 131072;
  float* xL     = x0 + 131072;
  float* fpart  = xL + 131072;              // 8*16*64 = 8192
  float* scores = fpart + 8192;             // 512
  float* invA   = scores + 512;             // 64
  float* biasA  = invA + 64;                // 64
  _Float16* wh  = (_Float16*)(biasA + 64);  // 36864 halfs (16B aligned)
                                            // total ws ~1.7 MB

  k_wprep   <<<dim3((C_ * KDIM + 255) / 256), dim3(256), 0, stream>>>(wv, wh);
  k_colstats<<<dim3(B_ * C_), dim3(256), 0, stream>>>(x, Sx, x0, xL);
  k_partial <<<dim3(16, B_),  dim3(256), 0, stream>>>(Sx, x0, xL, wq, wk, fpart);
  k_softmax <<<dim3(B_),      dim3(64),  0, stream>>>(fpart, gamma, beta, rmean, rvar,
                                                      scores, invA, biasA);
  const size_t lds_bytes = (size_t)(3 * XROW * XPITCH) * 2;   // 56160 B
  k_conv_fused<<<dim3(2, H_, B_), dim3(256), lds_bytes, stream>>>(
      x, wh, scores, invA, biasA, out);
}
